// MultiHeadGATLayer_89326729822277
// MI455X (gfx1250) — compile-verified, matrix operand order checked by reading, not run on hardware
//
#include <hip/hip_runtime.h>
#include <hip/hip_bf16.h>
#include <hip/hip_fp16.h>

typedef __attribute__((ext_vector_type(16))) _Float16 v16h;
typedef __attribute__((ext_vector_type(8)))  _Float16 v8h;
typedef __attribute__((ext_vector_type(8)))  float    v8f;

#define Bsz 8
#define Tsz 128
#define Nsz 64
#define Fsz 256
#define Hh  8
#define HD  32

// ---------------------------------------------------------------------------
// Kernel 0: repack W (f32 [out=256][in=256], torch layout) into f16 WMMA
// B-fragment order: wPack[((kb*16 + tn)*32 + lane)*16 + j]
//   j -> vgpr v = j/2, half = j%2
//   k  = kb*32 + (lane/16)*8 + (v%4)*2 + (v/4)*16 + half     (K index, = "in")
//   o  = tn*16 + (lane%16)                                   (N index, = "out")
// Then each lane of a wave loads its 16 halfs (32 B) contiguously.
// ---------------------------------------------------------------------------
__global__ void gat_pack_w(const float* __restrict__ W, _Float16* __restrict__ wPack) {
    int gid  = blockIdx.x * blockDim.x + threadIdx.x;   // 0 .. 65535
    int j    = gid & 15;
    int lane = (gid >> 4) & 31;
    int tn   = (gid >> 9) & 15;
    int kb   = gid >> 13;
    int v    = j >> 1;
    int hlf  = j & 1;
    int klane = ((lane >> 4) << 3) + ((v & 3) << 1) + ((v >> 2) << 4) + hlf;
    int k    = kb * 32 + klane;
    int o    = tn * 16 + (lane & 15);
    wPack[gid] = (_Float16)W[o * Fsz + k];
}

// ---------------------------------------------------------------------------
// LDS layout (static, 45568 B total; hF16 and WhT are unioned in time):
//   [0      .. 32768)  hF16 [64][256]  (stage 1 A operand)   |  WhT [256][64]
//   [32768  .. 40960)  attn [64][64]   f16 (per-head)
//   [40960  .. 43008)  sI   [64][8]    f32
//   [43008  .. 45056)  sJ   [64][8]    f32
//   [45056  .. 45568)  adjBits[64]     u64 row masks
// ---------------------------------------------------------------------------
__device__ __forceinline__ v16h frag_lds(const _Float16* p) {
    v8h lo = *(const v8h*)p;          // k = base_k + 0..7
    v8h hi = *(const v8h*)(p + 16);   // k = base_k + 16..23
    return __builtin_shufflevector(lo, hi, 0,1,2,3,4,5,6,7,8,9,10,11,12,13,14,15);
}

__global__ void __launch_bounds__(256)
gat_fused(const float* __restrict__ h_in, const int* __restrict__ adj,
          const float* __restrict__ a_vec, const _Float16* __restrict__ wPack,
          float* __restrict__ out) {
    __shared__ __align__(16) char smem[45568];
    _Float16* hF16 = (_Float16*)smem;                       // stage 1 only
    _Float16* WhT  = (_Float16*)smem;                       // after stage 1
    _Float16* attnB = (_Float16*)(smem + 32768);
    float*    sI   = (float*)(smem + 40960);
    float*    sJ   = (float*)(smem + 43008);
    unsigned long long* adjBits = (unsigned long long*)(smem + 45056);

    const int bt   = blockIdx.x;
    const int tid  = threadIdx.x;
    const int lane = tid & 31;
    const int wave = tid >> 5;
    const float* hBase = h_in + (size_t)bt * (Nsz * Fsz);

    // ---- stage 0: adjacency bitmasks + h tile -> f16 LDS --------------------
    if (tid < 64) {
        unsigned long long m = 0ull;
        #pragma unroll 8
        for (int j = 0; j < 64; ++j)
            if (adj[tid * 64 + j] != 0) m |= (1ull << j);
        adjBits[tid] = m;
    }
    #pragma unroll
    for (int r = 0; r < 16; ++r) {
        int idx = tid + r * 256;                 // float4 index, 0..4095
        float4 val = ((const float4*)hBase)[idx];
        union { _Float16 f[4]; uint2 u; } cv;
        cv.f[0] = (_Float16)val.x; cv.f[1] = (_Float16)val.y;
        cv.f[2] = (_Float16)val.z; cv.f[3] = (_Float16)val.w;
        ((uint2*)hF16)[idx] = cv.u;
    }
    __syncthreads();

    // ---- stage 1: Wh = h @ W^T  (64x256 = 4 Mtiles x 16 Ntiles, K=256) ------
    // wave w owns N-tiles {2w, 2w+1}, all 4 M-tiles -> 8 accum tiles.
    v8f acc[4][2];
    #pragma unroll
    for (int mt = 0; mt < 4; ++mt)
        #pragma unroll
        for (int ntl = 0; ntl < 2; ++ntl)
            acc[mt][ntl] = (v8f){0.f,0.f,0.f,0.f,0.f,0.f,0.f,0.f};

    const int laneM = lane & 15;
    const int laneK = (lane >> 4) << 3;
    for (int kb = 0; kb < 8; ++kb) {
        const int base_k = kb * 32 + laneK;
        v16h afr[4];
        #pragma unroll
        for (int mt = 0; mt < 4; ++mt)
            afr[mt] = frag_lds(&hF16[(mt * 16 + laneM) * Fsz + base_k]);
        v16h bfr[2];
        #pragma unroll
        for (int ntl = 0; ntl < 2; ++ntl) {
            int tn = wave * 2 + ntl;
            bfr[ntl] = *(const v16h*)&wPack[((kb * 16 + tn) * 32 + lane) * 16];
        }
        #pragma unroll
        for (int mt = 0; mt < 4; ++mt)
            #pragma unroll
            for (int ntl = 0; ntl < 2; ++ntl)
                acc[mt][ntl] = __builtin_amdgcn_wmma_f32_16x16x32_f16(
                    false, afr[mt], false, bfr[ntl],
                    (short)0, acc[mt][ntl], false, false);
    }
    __syncthreads();   // all hF16 reads done; region becomes WhT

    // ---- stage 1 epilogue: accumulators -> WhT f16 (column-major Wh) --------
    #pragma unroll
    for (int mt = 0; mt < 4; ++mt) {
        #pragma unroll
        for (int ntl = 0; ntl < 2; ++ntl) {
            int c = (wave * 2 + ntl) * 16 + laneM;   // feature column
            int mb = mt * 16 + laneK;                // node row base
            union { _Float16 f[8]; uint4 u; } cv;
            #pragma unroll
            for (int g = 0; g < 8; ++g) cv.f[g] = (_Float16)acc[mt][ntl][g];
            *(uint4*)&WhT[c * 64 + mb] = cv.u;       // one ds_store_b128
        }
    }
    __syncthreads();

    // ---- stage 2a: per-(node,head) scores s_i, s_j --------------------------
    #pragma unroll
    for (int r = 0; r < 2; ++r) {
        int task = tid + r * 256;        // 512 = 64 nodes x 8 heads
        int node = task & 63;
        int hh   = task >> 6;
        float si = 0.f, sj = 0.f;
        #pragma unroll 8
        for (int d = 0; d < HD; ++d) {
            float w = (float)WhT[(hh * HD + d) * 64 + node];
            si += w * a_vec[hh * 64 + d];
            sj += w * a_vec[hh * 64 + HD + d];
        }
        sI[node * Hh + hh] = si;
        sJ[node * Hh + hh] = sj;
    }
    __syncthreads();

    // ---- per-head: masked softmax + (attn @ Wh) + residual ELU --------------
    const int row  = tid >> 2;     // node i (softmax): 4 lanes per row
    const int part = tid & 3;      // j-range [16*part, 16*part+16)
    const int mt3  = wave >> 1;    // stage-3 tile assignment
    const int ntl3 = wave & 1;

    for (int hh = 0; hh < Hh; ++hh) {
        // softmax row (i = row) over neighbors j, 16 per lane
        unsigned long long m = adjBits[row];
        float si = sI[row * Hh + hh];
        float e[16];
        float mx = -__builtin_inff();
        #pragma unroll
        for (int q = 0; q < 16; ++q) {
            int j = part * 16 + q;
            float v;
            if ((m >> j) & 1ull) {
                v = si + sJ[j * Hh + hh];
                v = (v > 0.f) ? v : 0.2f * v;        // leaky_relu
            } else {
                v = -__builtin_inff();
            }
            e[q] = v;
            mx = fmaxf(mx, v);
        }
        mx = fmaxf(mx, __shfl_xor(mx, 1));
        mx = fmaxf(mx, __shfl_xor(mx, 2));
        float s = 0.f;
        #pragma unroll
        for (int q = 0; q < 16; ++q) {
            float t = (e[q] > -1e30f) ? __expf(e[q] - mx) : 0.f;
            e[q] = t; s += t;
        }
        s += __shfl_xor(s, 1);
        s += __shfl_xor(s, 2);
        float inv = (s > 0.f) ? (1.f / s) : 0.f;     // fully-masked row -> 0
        union { _Float16 f[16]; uint4 u[2]; } cv;
        #pragma unroll
        for (int q = 0; q < 16; ++q) cv.f[q] = (_Float16)(e[q] * inv);
        *(uint4*)&attnB[row * 64 + part * 16]     = cv.u[0];
        *(uint4*)&attnB[row * 64 + part * 16 + 8] = cv.u[1];
        __syncthreads();

        // stage 3: h_prime(:,head) = attn(64x64) @ Wh_head(64x32); 1 tile/wave
        v8f ao = (v8f){0.f,0.f,0.f,0.f,0.f,0.f,0.f,0.f};
        #pragma unroll
        for (int kb = 0; kb < 2; ++kb) {
            int base_k = kb * 32 + laneK;            // neighbor-j block
            v16h af = frag_lds(&attnB[(mt3 * 16 + laneM) * 64 + base_k]);
            int c = hh * HD + ntl3 * 16 + laneM;     // output feature column
            v16h bf = frag_lds(&WhT[c * 64 + base_k]);
            ao = __builtin_amdgcn_wmma_f32_16x16x32_f16(
                false, af, false, bf, (short)0, ao, false, false);
        }
        // fused residual + ELU, straight to HBM
        int ib = mt3 * 16 + laneK;
        int c  = hh * HD + ntl3 * 16 + laneM;
        #pragma unroll
        for (int g = 0; g < 8; ++g) {
            int i = ib + g;
            float x = ao[g] + hBase[i * Fsz + c];
            out[(size_t)bt * (Nsz * Fsz) + i * Fsz + c] =
                (x > 0.f) ? x : (__expf(x) - 1.f);
        }
        __syncthreads();   // attnB reused next head
    }
}

// ---------------------------------------------------------------------------
extern "C" void kernel_launch(void* const* d_in, const int* in_sizes, int n_in,
                              void* d_out, int out_size, void* d_ws, size_t ws_size,
                              hipStream_t stream) {
    const float* h   = (const float*)d_in[0];   // (8,128,64,256) f32
    const int*   adj = (const int*)d_in[1];     // (64,64) i32
    const float* W   = (const float*)d_in[2];   // (256,256) f32
    const float* a   = (const float*)d_in[3];   // (8,64) f32
    float* out = (float*)d_out;
    _Float16* wPack = (_Float16*)d_ws;          // 128 KB f16 packed W

    gat_pack_w<<<256, 256, 0, stream>>>(W, wPack);
    gat_fused<<<Bsz * Tsz, 256, 0, stream>>>(h, adj, a, wPack, out);
}
